// AutoregressiveFlowLayer_37881611550824
// MI455X (gfx1250) — matrix operand order repaired and checked by reading, not compiled, not run on hardware
//
#include <hip/hip_runtime.h>
#include <hip/hip_bf16.h>
#include <stdint.h>

typedef _Float16 f16;
typedef __attribute__((ext_vector_type(16))) _Float16 v16h;
typedef __attribute__((ext_vector_type(8)))  float    v8f;

#define R_ 8
#define B_ 16
#define D_ 32
#define H_ 128
#define N_ 2048
#define F_ 256
#define O_ 64          // 2*D
#define RB 128         // R*B

#define KP1 34         // padded K-stride (f16) for W1t rows (K=32)
#define KP2 130        // padded K-stride for W2t/W3t rows (K=128)
#define HSP 136        // padded row stride for h staging (K dir = 128)
#define XSP 34         // padded row stride for x staging (K dir = 32)
#define OSP 66         // padded row stride for out staging (64 cols)

#define NWAVE 4
#define THREADS (NWAVE*32)
#define NCHUNK 4
#define ROWS_PER_CHUNK (N_/NCHUNK)            // 512
#define TILES_PER_CHUNK (ROWS_PER_CHUNK/16)   // 32

#define HALF_LOG_2PI 0.9189385332046727f

union FragH { v16h v; uint32_t u[8]; };

// Wave-local LDS producer->consumer fence (per-wave staging buffers only).
__device__ __forceinline__ void wave_lds_fence() {
  __builtin_amdgcn_wave_barrier();
  asm volatile("s_wait_dscnt 0" ::: "memory");
  __builtin_amdgcn_wave_barrier();
}

// 16-bit A-matrix fragment (16x32 tile): lane half0 holds K=0..15,
// half1 holds K=16..31 -> 8 consecutive dwords starting at 8*half of the row.
__device__ __forceinline__ v16h load_fragA(const f16* row, int half) {
  const uint32_t* p = (const uint32_t*)row;
  FragH f; const int o = half * 8;
#pragma unroll
  for (int i = 0; i < 8; ++i) f.u[i] = p[o + i];
  return f.v;
}

// 16-bit B-matrix fragment (32x16 tile), rows stored [N][K]:
// lanes 0-15 hold K=0..15, lanes 16-31 hold K=16..31 -> 8 consecutive dwords.
__device__ __forceinline__ v16h load_fragB(const f16* row, int half) {
  const uint32_t* p = (const uint32_t*)row;
  FragH f; const int o = half * 8;
#pragma unroll
  for (int i = 0; i < 8; ++i) f.u[i] = p[o + i];
  return f.v;
}

__device__ __forceinline__ v8f wmma_f16(v16h a, v16h b, v8f c) {
  return __builtin_amdgcn_wmma_f32_16x16x32_f16(false, a, false, b,
                                                (short)0, c, false, false);
}

__global__ __launch_bounds__(THREADS, 1) void maf_kernel(
    const float* __restrict__ x,  const float* __restrict__ W1,
    const float* __restrict__ W2, const float* __restrict__ W3,
    const float* __restrict__ M1, const float* __restrict__ M2,
    const float* __restrict__ M3, const int* __restrict__ region_idx,
    float* __restrict__ out)
{
  __shared__ f16   sW1t[H_][KP1];          // (W1*M1)^T  [n][k]
  __shared__ f16   sW2t[H_][KP2];          // (W2*M2)^T
  __shared__ f16   sW3t[O_][KP2];          // (W3*M3)^T
  __shared__ f16   sXh[NWAVE][16][XSP];    // per-wave x tile (f16)
  __shared__ float sXf[NWAVE][16][D_];     // per-wave x tile (f32, for u)
  __shared__ f16   sHa[NWAVE][16][HSP];    // h1 staging
  __shared__ f16   sHb[NWAVE][16][HSP];    // h2 staging
  __shared__ float sOut[NWAVE][16][OSP];   // layer-3 output staging
  __shared__ float sRed[NWAVE][32];        // ll reduction

  const int g     = blockIdx.x & (RB - 1); // network id 0..127
  const int chunk = blockIdx.x >> 7;       // row chunk 0..3
  const int r = g >> 4, b = g & 15;
  const int tid = threadIdx.x;

  const float* w1 = W1 + (size_t)g * D_ * H_;
  const float* m1 = M1 + (size_t)g * D_ * H_;
  const float* w2 = W2 + (size_t)g * H_ * H_;
  const float* m2 = M2 + (size_t)g * H_ * H_;
  const float* w3 = W3 + (size_t)g * H_ * O_;
  const float* m3 = M3 + (size_t)g * H_ * O_;

  // Mask + convert + transpose weights into LDS (cooperative, once per block).
  for (int i = tid; i < D_ * H_; i += THREADS) {
    int n = i >> 5, k = i & 31; int s = k * H_ + n;
    sW1t[n][k] = (f16)(w1[s] * m1[s]);
  }
  for (int i = tid; i < H_ * H_; i += THREADS) {
    int n = i >> 7, k = i & 127; int s = k * H_ + n;
    sW2t[n][k] = (f16)(w2[s] * m2[s]);
  }
  for (int i = tid; i < O_ * H_; i += THREADS) {
    int n = i >> 7, k = i & 127; int s = k * O_ + n;
    sW3t[n][k] = (f16)(w3[s] * m3[s]);
  }
  __syncthreads();

  const int wave = tid >> 5, lane = tid & 31;
  const int mrow = lane & 15, half = lane >> 4;

  for (int t = wave; t < TILES_PER_CHUNK; t += NWAVE) {
    const int n0 = chunk * ROWS_PER_CHUNK + t * 16;

    // Gather X tile [16 rows x 32 region-features].
#pragma unroll
    for (int j = 0; j < 16; ++j) {
      int idx = lane * 16 + j;            // 0..511
      int rr = idx >> 5, cc = idx & 31;
      float v = x[(size_t)(n0 + rr) * F_ + region_idx[r * D_ + cc]];
      sXf[wave][rr][cc] = v;
      sXh[wave][rr][cc] = (f16)v;
    }
    wave_lds_fence();

    // ---- layer 1: h1 = relu(X[16x32] @ W1m[32x128]) ----
    v16h a1 = load_fragA(&sXh[wave][mrow][0], half);
#pragma unroll
    for (int nt = 0; nt < 8; ++nt) {
      v16h bf = load_fragB(&sW1t[nt * 16 + mrow][0], half);
      v8f c = {};
      c = wmma_f16(a1, bf, c);
#pragma unroll
      for (int i = 0; i < 8; ++i) {
        float vv = c[i]; vv = vv > 0.f ? vv : 0.f;
        sHa[wave][i + 8 * half][nt * 16 + mrow] = (f16)vv;
      }
    }
    wave_lds_fence();

    // ---- layer 2: h2 = relu(h1[16x128] @ W2m[128x128]) ----
#pragma unroll
    for (int nt = 0; nt < 8; ++nt) {
      v8f c = {};
#pragma unroll
      for (int kt = 0; kt < 4; ++kt) {
        v16h aa = load_fragA(&sHa[wave][mrow][kt * 32], half);
        v16h bb = load_fragB(&sW2t[nt * 16 + mrow][kt * 32], half);
        c = wmma_f16(aa, bb, c);
      }
#pragma unroll
      for (int i = 0; i < 8; ++i) {
        float vv = c[i]; vv = vv > 0.f ? vv : 0.f;
        sHb[wave][i + 8 * half][nt * 16 + mrow] = (f16)vv;
      }
    }
    wave_lds_fence();

    // ---- layer 3: out = h2[16x128] @ W3m[128x64] ----
#pragma unroll
    for (int nt = 0; nt < 4; ++nt) {
      v8f c = {};
#pragma unroll
      for (int kt = 0; kt < 4; ++kt) {
        v16h aa = load_fragA(&sHb[wave][mrow][kt * 32], half);
        v16h bb = load_fragB(&sW3t[nt * 16 + mrow][kt * 32], half);
        c = wmma_f16(aa, bb, c);
      }
#pragma unroll
      for (int i = 0; i < 8; ++i)
        sOut[wave][i + 8 * half][nt * 16 + mrow] = c[i];
    }
    wave_lds_fence();

    // ---- MAF inverse + log-prob: each lane handles row mrow, 16 d's ----
    float partial = 0.f;
#pragma unroll
    for (int dd = 0; dd < 16; ++dd) {
      int d = half * 16 + dd;
      float sh = sOut[wave][mrow][2 * d];
      float ls = sOut[wave][mrow][2 * d + 1];
      float xv = sXf[wave][mrow][d];
      float u = (xv - sh) * __expf(-ls);
      partial += fmaf(-0.5f * u, u, -HALF_LOG_2PI - ls);
    }
    sRed[wave][lane] = partial;
    wave_lds_fence();
    if (lane < 16) {
      float tot = sRed[wave][lane] + sRed[wave][lane + 16];
      out[(size_t)(n0 + lane) * RB + r * B_ + b] = tot;  // [N, R, B]
    }
    wave_lds_fence();  // protect staging reuse next iteration
  }
}

extern "C" void kernel_launch(void* const* d_in, const int* in_sizes, int n_in,
                              void* d_out, int out_size, void* d_ws, size_t ws_size,
                              hipStream_t stream) {
  const float* x  = (const float*)d_in[0];
  const float* W1 = (const float*)d_in[1];
  const float* W2 = (const float*)d_in[2];
  const float* W3 = (const float*)d_in[3];
  const float* M1 = (const float*)d_in[4];
  const float* M2 = (const float*)d_in[5];
  const float* M3 = (const float*)d_in[6];
  const int* ridx = (const int*)d_in[7];
  float* out = (float*)d_out;
  (void)in_sizes; (void)n_in; (void)out_size; (void)d_ws; (void)ws_size;

  maf_kernel<<<dim3(RB * NCHUNK), dim3(THREADS), 0, stream>>>(
      x, W1, W2, W3, M1, M2, M3, ridx, out);
}
